// GCN_norm_31104153158259
// MI455X (gfx1250) — matrix operand order repaired and checked by reading, not compiled, not run on hardware
//
#include <hip/hip_runtime.h>
#include <hip/hip_bf16.h>

// ---------------------------------------------------------------------------
// GCN forward for MI455X (gfx1250): bf16 WMMA GEMMs + L2-resident atomic SPMM
// ---------------------------------------------------------------------------

#define N_FEAT0 100
#define KPAD1   128

typedef __attribute__((ext_vector_type(16))) __bf16        v16bf;
typedef __attribute__((ext_vector_type(8)))  float         v8f;
typedef __attribute__((ext_vector_type(4)))  unsigned int  v4u;

union Frag16 { v16bf bf; v4u u4[2]; };

__device__ __forceinline__ unsigned short f2bf_bits(float f) {
    union { float f; unsigned u; } v; v.f = f;
    unsigned u = v.u;
    u += 0x7FFFu + ((u >> 16) & 1u);          // round-to-nearest-even
    return (unsigned short)(u >> 16);
}

// ---------------------------------------------------------------------------
// 1) clamp to [-1.8,1.8] + LayerNorm(100) -> bf16, K zero-padded to 128
//    one wave32 per row
// ---------------------------------------------------------------------------
__global__ __launch_bounds__(256) void ln_clamp_kernel(
        const float* __restrict__ x,
        const float* __restrict__ gamma, const float* __restrict__ beta,
        unsigned short* __restrict__ A, int M)
{
    int wave = threadIdx.x >> 5;
    int lane = threadIdx.x & 31;
    int row  = blockIdx.x * 8 + wave;
    if (row >= M) return;

    const float* xr = x + (long long)row * N_FEAT0;
    float v[4];
    float s = 0.f, s2 = 0.f;
#pragma unroll
    for (int i = 0; i < 4; ++i) {
        int j = lane + 32 * i;
        float t = (j < N_FEAT0) ? xr[j] : 0.f;
        t = fminf(fmaxf(t, -1.8f), 1.8f);
        v[i] = t;
        if (j < N_FEAT0) { s += t; s2 += t * t; }
    }
#pragma unroll
    for (int off = 16; off > 0; off >>= 1) {
        s  += __shfl_xor(s,  off, 32);
        s2 += __shfl_xor(s2, off, 32);
    }
    const float inv_n = 1.0f / (float)N_FEAT0;
    float mu   = s * inv_n;
    float var  = s2 * inv_n - mu * mu;
    float rstd = rsqrtf(var + 1e-5f);

    unsigned short* ar = A + (long long)row * KPAD1;
#pragma unroll
    for (int i = 0; i < 4; ++i) {
        int j = lane + 32 * i;                 // covers 0..127
        float y = 0.f;
        if (j < N_FEAT0) y = (v[i] - mu) * rstd * gamma[j] + beta[j];
        ar[j] = f2bf_bits(y);
    }
}

// ---------------------------------------------------------------------------
// 2) GEMM: Y[M,NPAD] = bf16(A[M,KPAD]) @ bf16(W[KV,NV]) + bias
//    v_wmma_f32_16x16x32_bf16; 8 waves/block, 16 rows per wave, 128 rows/block
// ---------------------------------------------------------------------------
template<int KPAD, int NPAD, int KV, int NV>
__global__ __launch_bounds__(256) void gemm_bf16_wmma(
        const unsigned short* __restrict__ A, int M,
        const float* __restrict__ W,      // [KV, NV] row-major fp32
        const float* __restrict__ bias,   // [NV] fp32
        float* __restrict__ Y)            // [M, NPAD] fp32
{
    __shared__ unsigned short sW[NPAD * KPAD];   // transposed: [n][k], bf16

    int tid = threadIdx.x;
    for (int idx = tid; idx < NPAD * KPAD; idx += 256) {
        int n = idx / KPAD;
        int k = idx - n * KPAD;
        float w = (n < NV && k < KV) ? W[k * NV + n] : 0.f;
        sW[idx] = f2bf_bits(w);
    }
    __syncthreads();

    int wave    = tid >> 5;
    int lane    = tid & 31;
    int lane15  = lane & 15;
    int laneHi  = lane >> 4;                     // 0 or 1
    int rowTile = blockIdx.x * 128 + wave * 16;

    int arow = rowTile + lane15;
    if (arow > M - 1) arow = M - 1;              // clamp: keep EXEC uniform
    const unsigned short* aBase = A + (long long)arow * KPAD + laneHi * 8;

    constexpr int NT = NPAD / 16;
    v8f acc[NT];
#pragma unroll
    for (int t = 0; t < NT; ++t)
#pragma unroll
        for (int i = 0; i < 8; ++i) acc[t][i] = 0.f;

#pragma unroll
    for (int k0 = 0; k0 < KPAD; k0 += 32) {
        // A fragment: lanes 0-15 K=[k0..k0+7],[k0+16..k0+23]; lanes 16-31 +8
        Frag16 a;
        const v4u* ap = (const v4u*)(aBase + k0);
        a.u4[0] = ap[0];
        a.u4[1] = ap[2];
#pragma unroll
        for (int t = 0; t < NT; ++t) {
            // B fragment: lane holds column n, K=[k0..k0+7],[k0+16..k0+23] (lanes>=16: +8)
            Frag16 b;
            const v4u* bp = (const v4u*)(&sW[(t * 16 + lane15) * KPAD + k0 + laneHi * 8]);
            b.u4[0] = bp[0];
            b.u4[1] = bp[2];
            acc[t] = __builtin_amdgcn_wmma_f32_16x16x32_bf16(
                false, a.bf, false, b.bf, (short)0, acc[t], false, false);
        }
    }

    // C/D layout: VGPR v -> M = v + 8*laneHi ; N = lane15
#pragma unroll
    for (int t = 0; t < NT; ++t) {
        int n = t * 16 + lane15;
        float bn = (n < NV) ? bias[n] : 0.f;
#pragma unroll
        for (int v = 0; v < 8; ++v) {
            int r = rowTile + v + laneHi * 8;
            if (r < M) Y[(long long)r * NPAD + n] = acc[t][v] + bn;
        }
    }
}

// ---------------------------------------------------------------------------
// 3) SPMM: Z[row[e], :] += vals[e] * Y[col[e], :]   (atomic f32, L2-resident)
//    one thread per (edge, 4-feature chunk)
// ---------------------------------------------------------------------------
__global__ __launch_bounds__(256) void spmm_kernel(
        const float* __restrict__ vals,
        const int* __restrict__ rows, const int* __restrict__ cols,
        const float* __restrict__ Y, int ldy,
        float* __restrict__ Z, int ldz, int Dv, int chunks, int E)
{
    long long t = (long long)blockIdx.x * blockDim.x + threadIdx.x;
    long long total = (long long)E * chunks;
    if (t >= total) return;
    int e = (int)(t / chunks);
    int c = (int)(t - (long long)e * chunks);
    int j = c * 4;

    float v  = vals[e];
    int   cn = cols[e];
    int   rn = rows[e];

    float4 y4 = *(const float4*)(Y + (long long)cn * ldy + j);
    float* zp = Z + (long long)rn * ldz + j;
    if (j + 3 < Dv) {
        atomicAdd(zp + 0, v * y4.x);
        atomicAdd(zp + 1, v * y4.y);
        atomicAdd(zp + 2, v * y4.z);
        atomicAdd(zp + 3, v * y4.w);
    } else {
        if (j + 0 < Dv) atomicAdd(zp + 0, v * y4.x);
        if (j + 1 < Dv) atomicAdd(zp + 1, v * y4.y);
        if (j + 2 < Dv) atomicAdd(zp + 2, v * y4.z);
        if (j + 3 < Dv) atomicAdd(zp + 3, v * y4.w);
    }
}

// ---------------------------------------------------------------------------
// 4) ELU + fp32 -> bf16 conversion (next GEMM's A operand)
// ---------------------------------------------------------------------------
__global__ __launch_bounds__(256) void elu_to_bf16_kernel(
        const float* __restrict__ Z, unsigned short* __restrict__ A, int n)
{
    int i = blockIdx.x * 256 + threadIdx.x;
    if (i >= n) return;
    float x = Z[i];
    float y = (x > 0.f) ? x : expm1f(x);
    A[i] = f2bf_bits(y);
}

__global__ __launch_bounds__(256) void zero_f32_kernel(float* __restrict__ p, int n)
{
    int stride = gridDim.x * 256;
    for (int i = blockIdx.x * 256 + threadIdx.x; i < n; i += stride) p[i] = 0.f;
}

// ---------------------------------------------------------------------------
// launcher
// ---------------------------------------------------------------------------
extern "C" void kernel_launch(void* const* d_in, const int* in_sizes, int n_in,
                              void* d_out, int out_size, void* d_ws, size_t ws_size,
                              hipStream_t stream)
{
    const float* x        = (const float*)d_in[0];
    const float* adj_vals = (const float*)d_in[1];
    const int*   adj_row  = (const int*)  d_in[2];
    const int*   adj_col  = (const int*)  d_in[3];
    const float* gamma    = (const float*)d_in[4];
    const float* beta     = (const float*)d_in[5];
    const float* W1       = (const float*)d_in[6];
    const float* b1       = (const float*)d_in[7];
    const float* W2       = (const float*)d_in[8];
    const float* b2       = (const float*)d_in[9];
    const float* W3       = (const float*)d_in[10];
    const float* b3       = (const float*)d_in[11];

    const int M = in_sizes[0] / N_FEAT0;   // 100000
    const int E = in_sizes[1];             // 1600000

    char* ws = (char*)d_ws;
    size_t offA = 0;                                                  // bf16 acts
    size_t offY = offA + (size_t)M * KPAD1 * sizeof(unsigned short);  // gemm out
    size_t offZ = offY + (size_t)M * 64 * sizeof(float);              // spmm accum
    unsigned short* Abf = (unsigned short*)(ws + offA);
    float* Y   = (float*)(ws + offY);
    float* Z   = (float*)(ws + offZ);
    float* out = (float*)d_out;

    dim3 blk(256);
    const int gGemm = (M + 127) / 128;
    const int nZ    = M * 64;

    // Layer 0: clamp + LayerNorm -> bf16 [M,128]
    ln_clamp_kernel<<<(M + 7) / 8, blk, 0, stream>>>(x, gamma, beta, Abf, M);

    // Layer 1: GEMM(100->64) -> spmm -> elu
    gemm_bf16_wmma<128, 64, 100, 64><<<gGemm, blk, 0, stream>>>(Abf, M, W1, b1, Y);
    zero_f32_kernel<<<2048, blk, 0, stream>>>(Z, nZ);
    {
        long long total = (long long)E * 16;
        spmm_kernel<<<(int)((total + 255) / 256), blk, 0, stream>>>(
            adj_vals, adj_row, adj_col, Y, 64, Z, 64, 64, 16, E);
    }
    elu_to_bf16_kernel<<<(nZ + 255) / 256, blk, 0, stream>>>(Z, Abf, nZ);

    // Layer 2: GEMM(64->64) -> spmm -> elu
    gemm_bf16_wmma<64, 64, 64, 64><<<gGemm, blk, 0, stream>>>(Abf, M, W2, b2, Y);
    zero_f32_kernel<<<2048, blk, 0, stream>>>(Z, nZ);
    {
        long long total = (long long)E * 16;
        spmm_kernel<<<(int)((total + 255) / 256), blk, 0, stream>>>(
            adj_vals, adj_row, adj_col, Y, 64, Z, 64, 64, 16, E);
    }
    elu_to_bf16_kernel<<<(nZ + 255) / 256, blk, 0, stream>>>(Z, Abf, nZ);

    // Layer 3: GEMM(64->47, N padded to 48) -> spmm into d_out
    gemm_bf16_wmma<64, 48, 64, 47><<<gGemm, blk, 0, stream>>>(Abf, M, W3, b3, Y);
    zero_f32_kernel<<<2048, blk, 0, stream>>>(out, M * 47);
    {
        long long total = (long long)E * 12;
        spmm_kernel<<<(int)((total + 255) / 256), blk, 0, stream>>>(
            adj_vals, adj_row, adj_col, Y, 48, out, 47, 47, 12, E);
    }
}
